// ExpertFFNUnique_9929964389207
// MI455X (gfx1250) — compile-verified, matrix operand order checked by reading, not run on hardware
//
#include <hip/hip_runtime.h>

typedef __bf16 v16bf __attribute__((ext_vector_type(16)));
typedef float  v8f   __attribute__((ext_vector_type(8)));

#define D_MODEL 768
#define D_FF    3072
#define NEXP    4
#define NTOK    16384     // 8 * 2048
#define M_TILE  64        // tokens per workgroup tile (4 m-tiles)
#define NWAVES  16        // 512 threads
#define FFC     256       // d_ff chunk per fused pass (16 n-tiles, 1/wave)
#define XSTR    (D_MODEL + 8)   // LDS stride (bf16) for X tile (16B-aligned rows)
#define HSTR    (FFC + 8)       // LDS stride (bf16) for H tile (16B-aligned rows)
#define KS1     (D_MODEL / 32)  // 24 k-steps for GEMM1
#define KS2     (D_FF / 32)     // 96 k-steps for GEMM2

// gfx1250 has a v_tanh_f32 transcendental; use it if clang exposes it.
__device__ __forceinline__ float fast_tanh(float u) {
#if __has_builtin(__builtin_amdgcn_tanhf)
  return __builtin_amdgcn_tanhf(u);
#else
  float e = __expf(-2.0f * __builtin_fabsf(u));
  float t = __fdividef(1.0f - e, 1.0f + e);
  return __builtin_copysignf(t, u);
#endif
}

__device__ __forceinline__ float gelu_tanh(float x) {
  float u = 0.7978845608028654f * (x + 0.044715f * x * x * x);
  return 0.5f * x * (1.0f + fast_tanh(u));
}

// ---------------- weight pre-pack: fp32 [K,N] -> bf16 WMMA-B fragments ------
// Packed layout: for n-tile nt, k-step ks (32 rows), lane l: 16 contiguous
// bf16 = B-fragment of that lane.  dst[((nt*(K/32)+ks)*32 + l)*16 + j]
//   col n = nt*16 + (l&15),  row k = ks*32 + (l>>4)*8 + (j<8 ? j : j+8)
__global__ __launch_bounds__(256) void moe_pack_b(
    const float* __restrict__ src, __bf16* __restrict__ dst, int K, int N) {
  const size_t per = (size_t)K * N;
  size_t o = (size_t)blockIdx.x * 256 + threadIdx.x;   // flat output index
  if (o >= per) return;
  const int e = blockIdx.y;
  int j    = (int)(o & 15);
  int lane = (int)((o >> 4) & 31);
  size_t blk = o >> 9;                 // (nt * K/32 + ks)
  int ks = (int)(blk % (K / 32));
  int nt = (int)(blk / (K / 32));
  int n = nt * 16 + (lane & 15);
  int k = ks * 32 + (lane >> 4) * 8 + (j < 8 ? j : j + 8);
  dst[e * per + o] = (__bf16)src[e * per + (size_t)k * N + n];
}

// ---------------- router: softmax(x @ Wg + bg), top-1 gate + route ----------
__global__ __launch_bounds__(256) void moe_router(
    const float* __restrict__ x, const float* __restrict__ Wg,
    const float* __restrict__ bg, float* __restrict__ gate,
    int* __restrict__ route) {
  int t    = blockIdx.x * 8 + (threadIdx.x >> 5);   // one wave32 per token
  int lane = threadIdx.x & 31;
  const float* xr = x + (size_t)t * D_MODEL;
  float s0 = 0.f, s1 = 0.f, s2 = 0.f, s3 = 0.f;
  for (int d = lane; d < D_MODEL; d += 32) {
    float xv = xr[d];
    s0 += xv * Wg[d * 4 + 0];
    s1 += xv * Wg[d * 4 + 1];
    s2 += xv * Wg[d * 4 + 2];
    s3 += xv * Wg[d * 4 + 3];
  }
  for (int off = 16; off > 0; off >>= 1) {
    s0 += __shfl_xor(s0, off, 32);
    s1 += __shfl_xor(s1, off, 32);
    s2 += __shfl_xor(s2, off, 32);
    s3 += __shfl_xor(s3, off, 32);
  }
  if (lane == 0) {
    float l[4] = { s0 + bg[0], s1 + bg[1], s2 + bg[2], s3 + bg[3] };
    int am = 0; float mx = l[0];
#pragma unroll
    for (int e = 1; e < 4; ++e) if (l[e] > mx) { mx = l[e]; am = e; }
    float den = 0.f;
#pragma unroll
    for (int e = 0; e < 4; ++e) den += __expf(l[e] - mx);
    gate[t]  = 1.0f / den;     // max prob = exp(0)/den
    route[t] = am;
  }
}

// ---------------- routing bookkeeping --------------------------------------
__global__ void moe_zero(int* __restrict__ counts) {
  if (threadIdx.x < NEXP) counts[threadIdx.x] = 0;
}

__global__ __launch_bounds__(256) void moe_hist(const int* __restrict__ route,
                                                int* __restrict__ counts) {
  int t = blockIdx.x * 256 + threadIdx.x;
  atomicAdd(&counts[route[t]], 1);
}

__global__ void moe_scan(const int* __restrict__ counts,
                         int* __restrict__ offsets, int* __restrict__ cursor) {
  if (threadIdx.x == 0) {
    int acc = 0;
    for (int e = 0; e < NEXP; ++e) { offsets[e] = acc; cursor[e] = acc; acc += counts[e]; }
  }
}

__global__ __launch_bounds__(256) void moe_scatter(const int* __restrict__ route,
                                                   int* __restrict__ cursor,
                                                   int* __restrict__ perm) {
  int t = blockIdx.x * 256 + threadIdx.x;
  int p = atomicAdd(&cursor[route[t]], 1);
  perm[p] = t;
}

// ---------------- fused expert FFN via bf16 WMMA ---------------------------
// grid = (NEXP, NTOK/M_TILE), block = 512 (16 wave32)
__global__ __launch_bounds__(512) void moe_ffn(
    const float* __restrict__ x,
    const __bf16* __restrict__ pW1, const float* __restrict__ b1,
    const __bf16* __restrict__ pW2, const float* __restrict__ b2,
    const float* __restrict__ gate, const int* __restrict__ perm,
    const int* __restrict__ counts, const int* __restrict__ offsets,
    float* __restrict__ out) {
  const int e    = blockIdx.x;
  const int tile = blockIdx.y;
  const int cnt  = counts[e];
  const int rowStart = tile * M_TILE;
  if (rowStart >= cnt) return;                    // uniform across block
  const int base = offsets[e];

  __shared__ __bf16 lds_x[M_TILE * XSTR];         //  99.3 KB
  __shared__ __bf16 lds_h[M_TILE * HSTR];         //  33.8 KB
  __shared__ int   rowTok[M_TILE];
  __shared__ float rowGate[M_TILE];

  const int tid  = threadIdx.x;
  const int wave = tid >> 5;      // 0..15
  const int lane = tid & 31;
  const int half = lane >> 4;     // K-half selector (wave32 WMMA layout)
  const int idx  = lane & 15;     // row (A) / col (B) / col (C) within tile

  if (tid < M_TILE) {
    int r = rowStart + tid;
    int tok = (r < cnt) ? perm[base + r] : -1;
    rowTok[tid]  = tok;
    rowGate[tid] = (tok >= 0) ? gate[tok] : 0.f;
  }
  __syncthreads();

  // gather X tile (64 x 768 fp32) -> bf16 LDS; float4 loads, coalesced
  for (int i = tid; i < M_TILE * (D_MODEL / 4); i += 512) {
    int r  = i / (D_MODEL / 4), c4 = i % (D_MODEL / 4);
    int tok = rowTok[r];
    float4 v = make_float4(0.f, 0.f, 0.f, 0.f);
    if (tok >= 0) v = ((const float4*)(x + (size_t)tok * D_MODEL))[c4];
    __bf16* dst = &lds_x[r * XSTR + c4 * 4];
    dst[0] = (__bf16)v.x; dst[1] = (__bf16)v.y;
    dst[2] = (__bf16)v.z; dst[3] = (__bf16)v.w;
  }
  __syncthreads();

  const __bf16* pW1e = pW1 + (size_t)e * D_MODEL * D_FF;
  const __bf16* pW2e = pW2 + (size_t)e * D_FF * D_MODEL;
  const float*  b1e  = b1 + (size_t)e * D_FF;
  const float*  b2e  = b2 + (size_t)e * D_MODEL;

  // stage-1: wave owns one 16-col n-tile of the FFC-wide H chunk, all 4 m-tiles
  const int nt1 = wave;                           // 0..15
  // stage-2: wave owns 48 output columns = 3 n-tiles x 4 m-tiles
  const int nBaseW = wave * (D_MODEL / NWAVES);   // wave * 48

  v8f acc[4][3];
#pragma unroll
  for (int m = 0; m < 4; ++m)
#pragma unroll
    for (int n = 0; n < 3; ++n) acc[m][n] = (v8f)0.0f;

  for (int fc = 0; fc < D_FF / FFC; ++fc) {
    const int ffBase = fc * FFC;

    // ---- stage 1: H = gelu(X @ W1[:, chunk] + b1[chunk]) ----
    // Packed B fragment: one 32B load per lane; reused across 4 m-tiles.
    v8f hacc[4];
#pragma unroll
    for (int m = 0; m < 4; ++m) hacc[m] = (v8f)0.0f;
    {
      const int ntGlob = (ffBase >> 4) + nt1;     // global n-tile in W1
      const __bf16* w1p = pW1e + (((size_t)ntGlob * KS1) * 32 + lane) * 16;
      for (int ks = 0; ks < KS1; ++ks, w1p += 512) {
        v16bf b = *(const v16bf*)w1p;
        const int k0 = ks * 32;
#pragma unroll
        for (int m = 0; m < 4; ++m) {
          v16bf a;
#pragma unroll
          for (int j = 0; j < 16; ++j) {
            int kk = k0 + half * 8 + (j < 8 ? j : j + 8);
            a[j] = lds_x[(m * 16 + idx) * XSTR + kk];
          }
          hacc[m] = __builtin_amdgcn_wmma_f32_16x16x32_bf16(false, a, false, b,
                                                            (short)0, hacc[m], false, false);
        }
      }
    }
    {
      const int   c    = nt1 * 16 + idx;          // col within chunk
      const float bias = b1e[ffBase + c];
#pragma unroll
      for (int m = 0; m < 4; ++m)
#pragma unroll
        for (int v = 0; v < 8; ++v) {
          int r = m * 16 + (half ? v + 8 : v);
          lds_h[r * HSTR + c] = (__bf16)gelu_tanh(hacc[m][v] + bias);
        }
    }
    __syncthreads();

    // ---- stage 2: OUT += H @ W2[chunk, :] ----
    for (int k0 = 0; k0 < FFC; k0 += 32) {
      const int ks = (ffBase + k0) >> 5;          // global k-step in W2
#pragma unroll
      for (int n = 0; n < 3; ++n) {
        const int ntGlob = wave * 3 + n;          // global n-tile in W2
        const __bf16* w2p = pW2e + (((size_t)ntGlob * KS2 + ks) * 32 + lane) * 16;
        v16bf b = *(const v16bf*)w2p;
#pragma unroll
        for (int m = 0; m < 4; ++m) {
          v16bf a;
#pragma unroll
          for (int j = 0; j < 16; ++j) {
            int kk = k0 + half * 8 + (j < 8 ? j : j + 8);
            a[j] = lds_h[(m * 16 + idx) * HSTR + kk];
          }
          acc[m][n] = __builtin_amdgcn_wmma_f32_16x16x32_bf16(false, a, false, b,
                                                              (short)0, acc[m][n], false, false);
        }
      }
    }
    __syncthreads();
  }

  // ---- epilogue: out[tok] = gate * (acc + b2) ----
#pragma unroll
  for (int m = 0; m < 4; ++m) {
#pragma unroll
    for (int v = 0; v < 8; ++v) {
      int r = m * 16 + (half ? v + 8 : v);
      int tok = rowTok[r];
      if (tok < 0) continue;
      float g = rowGate[r];
#pragma unroll
      for (int n = 0; n < 3; ++n) {
        int c = nBaseW + n * 16 + idx;
        out[(size_t)tok * D_MODEL + c] = g * (acc[m][n][v] + b2e[c]);
      }
    }
  }
}

// ---------------------------------------------------------------------------
extern "C" void kernel_launch(void* const* d_in, const int* in_sizes, int n_in,
                              void* d_out, int out_size, void* d_ws, size_t ws_size,
                              hipStream_t stream) {
  const float* x  = (const float*)d_in[0];
  const float* Wg = (const float*)d_in[1];
  const float* bg = (const float*)d_in[2];
  const float* W1 = (const float*)d_in[3];
  const float* b1 = (const float*)d_in[4];
  const float* W2 = (const float*)d_in[5];
  const float* b2 = (const float*)d_in[6];
  float* out = (float*)d_out;

  // workspace carve-up: 2 x 18.9 MB packed bf16 weights + ~196 KB routing
  const size_t WELEMS = (size_t)NEXP * D_MODEL * D_FF;
  char* p = (char*)d_ws;
  __bf16* pW1 = (__bf16*)p;  p += WELEMS * sizeof(__bf16);
  __bf16* pW2 = (__bf16*)p;  p += WELEMS * sizeof(__bf16);
  float* gate   = (float*)p; p += (size_t)NTOK * sizeof(float);
  int*   route  = (int*)p;   p += (size_t)NTOK * sizeof(int);
  int*   counts = (int*)p;   p += NEXP * sizeof(int);
  int*   offs   = (int*)p;   p += NEXP * sizeof(int);
  int*   cursor = (int*)p;   p += NEXP * sizeof(int);
  int*   perm   = (int*)p;

  const int packBlocks = (int)((D_MODEL * (size_t)D_FF + 255) / 256);
  moe_pack_b<<<dim3(packBlocks, NEXP), 256, 0, stream>>>(W1, pW1, D_MODEL, D_FF);
  moe_pack_b<<<dim3(packBlocks, NEXP), 256, 0, stream>>>(W2, pW2, D_FF, D_MODEL);

  moe_router <<<NTOK / 8,   256, 0, stream>>>(x, Wg, bg, gate, route);
  moe_zero   <<<1,          32,  0, stream>>>(counts);
  moe_hist   <<<NTOK / 256, 256, 0, stream>>>(route, counts);
  moe_scan   <<<1,          1,   0, stream>>>(counts, offs, cursor);
  moe_scatter<<<NTOK / 256, 256, 0, stream>>>(route, cursor, perm);

  dim3 grid(NEXP, NTOK / M_TILE);
  moe_ffn<<<grid, 512, 0, stream>>>(x, pW1, b1, pW2, b2, gate, perm, counts, offs, out);
}
